// PointNetSetAbstractionMsg_24575802868370
// MI455X (gfx1250) — matrix operand order repaired and checked by reading, not compiled, not run on hardware
//
#include <hip/hip_runtime.h>
#include <hip/hip_bf16.h>

typedef __attribute__((ext_vector_type(16))) _Float16 v16h;
typedef __attribute__((ext_vector_type(8)))  _Float16 v8h;
typedef __attribute__((ext_vector_type(8)))  float    v8f;

namespace {
constexpr int B  = 16;
constexpr int N  = 4096;
constexpr int D  = 64;
constexpr int S  = 512;
constexpr int CIN   = D + 3;   // 67
constexpr int K1PAD = 96;      // pad 67 -> 96 (multiple of WMMA K=32)
constexpr int ROWS  = 128;     // rows (b,s,k) per fused-MLP block
constexpr int LDX   = 104;     // LDS stride (halfs) for buffer X (>=96), 0 mod 8
constexpr int LDY   = 136;     // LDS stride (halfs) for buffer Y (>=128), 0 mod 8
constexpr int WMAX  = 96 * 128; // max weight tile (halfs)
}

// ---------------------------------------------------------------------------
// WMMA fragment loads, contiguous k order (B packed to match).
// A 16x32 f16 (MxK): lanes 0-15 row M=lane hold K 0-15;
//                    lanes 16-31 row M=lane-16 hold K 16-31.
// Vectorized: two aligned 16B LDS reads (ds_load_b128).
// ---------------------------------------------------------------------------
__device__ __forceinline__ v16h load_a_frag(const _Float16* src, int ld,
                                            int row0, int k0, int lane) {
  const int r  = row0 + (lane & 15);
  const int ko = (lane >> 4) * 16;
  const _Float16* p = src + r * ld + k0 + ko;   // 16B aligned: ld,k0,ko = 0 mod 8
  const v8h lo = *(const v8h*)(p);
  const v8h hi = *(const v8h*)(p + 8);
  v16h a;
#pragma unroll
  for (int j = 0; j < 8; ++j) { a[j] = lo[j]; a[8 + j] = hi[j]; }
  return a;
}

// B fragments are pre-swizzled into register layout: fragment (kk,ct) is 512
// contiguous halfs = [lane][16]; a fragment load is one aligned v16h LDS read.
__device__ __forceinline__ v16h load_b_frag_swz(const _Float16* wb, int frag,
                                                int lane) {
  return ((const v16h*)wb)[(frag << 5) + lane];
}

// ---------------------------------------------------------------------------
// Kernel 1: farthest point sampling, one block per batch.
// ---------------------------------------------------------------------------
__global__ __launch_bounds__(256) void fps_kernel(const float* __restrict__ xyz,
                                                  int* __restrict__ fps_idx,
                                                  float* __restrict__ newxyz_ws,
                                                  float* __restrict__ out_xyz) {
  const int b   = blockIdx.x;
  const int tid = threadIdx.x;
  __shared__ float xs[N], ys[N], zs[N], dist[N];
  __shared__ float rv[256];
  __shared__ int   ri[256];
  __shared__ int   s_cur;

  const float* px = xyz + (size_t)b * 3 * N;
  for (int n = tid; n < N; n += 256) {
    xs[n] = px[n];
    ys[n] = px[N + n];
    zs[n] = px[2 * N + n];
    dist[n] = 1e10f;
  }
  if (tid == 0) s_cur = 0;
  __syncthreads();

  for (int s = 0; s < S; ++s) {
    const int cur = s_cur;
    const float cx = xs[cur], cy = ys[cur], cz = zs[cur];
    if (tid == 0) {
      fps_idx[b * S + s] = cur;
      float* nw = newxyz_ws + ((size_t)b * S + s) * 3;
      nw[0] = cx; nw[1] = cy; nw[2] = cz;
      out_xyz[((size_t)b * 3 + 0) * S + s] = cx;
      out_xyz[((size_t)b * 3 + 1) * S + s] = cy;
      out_xyz[((size_t)b * 3 + 2) * S + s] = cz;
    }
    float best = -1.0f;
    int   bi   = 0;
    for (int n = tid; n < N; n += 256) {
      const float dx = xs[n] - cx, dy = ys[n] - cy, dz = zs[n] - cz;
      const float d  = dx * dx + dy * dy + dz * dz;
      const float dn = fminf(dist[n], d);
      dist[n] = dn;
      if (dn > best) { best = dn; bi = n; }   // strict > keeps lowest index
    }
    rv[tid] = best; ri[tid] = bi;
    __syncthreads();
    for (int off = 128; off > 0; off >>= 1) {
      if (tid < off) {
        const float ov = rv[tid + off];
        const int   oi = ri[tid + off];
        if (ov > rv[tid] || (ov == rv[tid] && oi < ri[tid])) { rv[tid] = ov; ri[tid] = oi; }
      }
      __syncthreads();
    }
    if (tid == 0) s_cur = ri[0];
    __syncthreads();
  }
}

// ---------------------------------------------------------------------------
// Kernel 2: ball query. One thread per (b,s); scan in index order, first K.
// ---------------------------------------------------------------------------
template <int K>
__global__ __launch_bounds__(256) void ballq_kernel(const float* __restrict__ xyz,
                                                    const float* __restrict__ newxyz,
                                                    int* __restrict__ gidx,
                                                    float r2) {
  const int t = blockIdx.x * 256 + threadIdx.x;
  if (t >= B * S) return;
  const int b = t / S;
  const float cx = newxyz[t * 3 + 0];
  const float cy = newxyz[t * 3 + 1];
  const float cz = newxyz[t * 3 + 2];
  const float* px = xyz + (size_t)b * 3 * N;
  int* out = gidx + (size_t)t * K;
  int cnt = 0;
  for (int n = 0; n < N; ++n) {
    const float dx = px[n] - cx, dy = px[N + n] - cy, dz = px[2 * N + n] - cz;
    if (dx * dx + dy * dy + dz * dz <= r2) {
      out[cnt++] = n;
      if (cnt == K) break;
    }
  }
  const int pad = (cnt > 0) ? out[0] : (N - 1);
  for (int k = cnt; k < K; ++k) out[k] = pad;
}

// ---------------------------------------------------------------------------
// Kernel 3: fold BN into scale/shift; pad + convert weights to f16 and
// pre-swizzle into WMMA B-fragment register layout:
//   fragment (kk = k/32, ct = c/16) -> 512 halfs: [lane][j]
//   lane = (kr/16)*16 + (c%16), j = kr%16, kr = k%32.
// ---------------------------------------------------------------------------
__global__ __launch_bounds__(256) void prep_layer_kernel(
    const float* __restrict__ w, const float* __restrict__ bias,
    const float* __restrict__ gamma, const float* __restrict__ beta,
    const float* __restrict__ mean, const float* __restrict__ var,
    int cin, int cout, int kpad,
    _Float16* __restrict__ wout, float* __restrict__ scale, float* __restrict__ shift) {
  const int t = blockIdx.x * 256 + threadIdx.x;
  if (t < cout) {
    const float sc = gamma[t] * rsqrtf(var[t] + 0.001f);
    scale[t] = sc;
    shift[t] = (bias[t] - mean[t]) * sc + beta[t];
  }
  const int nct = cout >> 4;
  for (int e = t; e < kpad * cout; e += gridDim.x * 256) {
    const int k = e / cout, c = e % cout;
    const int kk = k >> 5, kr = k & 31;
    const int ct = c >> 4, cl = c & 15;
    const int lane = ((kr >> 4) << 4) | cl;
    const int idx  = ((kk * nct + ct) << 9) + (lane << 4) + (kr & 15);
    wout[idx] = (_Float16)((k < cin) ? w[k * cout + c] : 0.0f);
  }
}

// ---------------------------------------------------------------------------
// One fused MLP layer: LDS(in) x LDS(weights, fragment layout) -> LDS(out).
// WMMA f16->f32, folded-BN scale/shift + ReLU epilogue.
// 8 wave32 per block; wave w owns rows 16w..16w+15.
// ---------------------------------------------------------------------------
template <int KDIM, int COUT, int LDIN, int LDOUT>
__device__ __forceinline__ void mlp_layer(const _Float16* in, _Float16* outb,
                                          _Float16* wb,
                                          const _Float16* __restrict__ wglob,
                                          const float* __restrict__ sc,
                                          const float* __restrict__ sh, int tid) {
  // Weight tile copy: already fragment-ordered, so a plain b128 memcpy.
  {
    const v8h* src = (const v8h*)wglob;
    v8h*       dst = (v8h*)wb;
#pragma unroll 4
    for (int e = tid; e < (KDIM * COUT) / 8; e += 256) dst[e] = src[e];
  }
  __syncthreads();
  const int lane = tid & 31;
  const int row0 = (tid >> 5) * 16;
#pragma unroll
  for (int ct = 0; ct < COUT / 16; ++ct) {
    v8f acc = {};
#pragma unroll
    for (int kk = 0; kk < KDIM / 32; ++kk) {
      v16h a  = load_a_frag(in, LDIN, row0, kk * 32, lane);
      v16h bb = load_b_frag_swz(wb, kk * (COUT / 16) + ct, lane);
      acc = __builtin_amdgcn_wmma_f32_16x16x32_f16(false, a, false, bb,
                                                   (short)0, acc, false, false);
    }
    const int   c     = ct * 16 + (lane & 15);
    const float scale = sc[c];
    const float shift = sh[c];
    const int   rbase = row0 + ((lane >> 4) << 3);
#pragma unroll
    for (int j = 0; j < 8; ++j) {
      float h = fmaf(acc[j], scale, shift);
      h = fmaxf(h, 0.0f);
      outb[(rbase + j) * LDOUT + c] = (_Float16)h;
    }
  }
  __syncthreads();
}

// ---------------------------------------------------------------------------
// Kernel 4: fused gather -> 3x(GEMM+BN+ReLU) -> max over K, per branch.
// Block = 256 threads (8 wave32), 128 rows = ROWS/K groups of (b,s).
// All activations stay in LDS (~90 KB of the 320 KB WGP pool).
// ---------------------------------------------------------------------------
template <int K, int C1, int C2, int C3>
__global__ __launch_bounds__(256) void msg_branch_kernel(
    const float* __restrict__ xyz, const float* __restrict__ points,
    const float* __restrict__ newxyz, const int* __restrict__ gidx,
    const _Float16* __restrict__ w1, const _Float16* __restrict__ w2,
    const _Float16* __restrict__ w3,
    const float* __restrict__ sc1, const float* __restrict__ sh1,
    const float* __restrict__ sc2, const float* __restrict__ sh2,
    const float* __restrict__ sc3, const float* __restrict__ sh3,
    float* __restrict__ out_points, int chofs) {
  constexpr int GROUPS = ROWS / K;
  __shared__ alignas(16) _Float16 bufX[ROWS * LDX];  // A0 (96 ch) -> H2 (<=96 ch)
  __shared__ alignas(16) _Float16 bufY[ROWS * LDY];  // H1 (<=64 ch) -> H3 (<=128 ch)
  __shared__ alignas(16) _Float16 wbuf[WMAX];        // fragment-ordered weights

  const int tid = threadIdx.x;
  const int g0  = blockIdx.x * GROUPS;    // global group = b*S + s

  __builtin_prefetch(w1, 0, 0);
  __builtin_prefetch(w2, 0, 0);
  __builtin_prefetch(w3, 0, 0);

  // Stage 0: gather concat(points[gidx], xyz[gidx]-new_xyz) into bufX as f16.
  for (int e = tid; e < ROWS * K1PAD; e += 256) {
    const int row = e / K1PAD, c = e % K1PAD;
    const int g = g0 + row / K;
    const int k = row % K;
    const int b = g / S;
    const int idx = gidx[(size_t)g * K + k];
    float v;
    if (c < D) {
      v = points[((size_t)b * D + c) * N + idx];
    } else if (c < CIN) {
      const int a = c - D;
      v = xyz[((size_t)b * 3 + a) * N + idx] - newxyz[(size_t)g * 3 + a];
    } else {
      v = 0.0f;
    }
    bufX[row * LDX + c] = (_Float16)v;
  }
  __syncthreads();

  mlp_layer<K1PAD, C1, LDX, LDY>(bufX, bufY, wbuf, w1, sc1, sh1, tid);
  mlp_layer<C1,    C2, LDY, LDX>(bufY, bufX, wbuf, w2, sc2, sh2, tid);
  mlp_layer<C2,    C3, LDX, LDY>(bufX, bufY, wbuf, w3, sc3, sh3, tid);

  // Max over K neighbors, write (B, 320, S) slice at channel offset chofs.
  for (int e = tid; e < GROUPS * C3; e += 256) {
    const int g = e / C3, c = e % C3;
    float m = -1e30f;
#pragma unroll 4
    for (int k = 0; k < K; ++k)
      m = fmaxf(m, (float)bufY[(g * K + k) * LDY + c]);
    const int gg = g0 + g;
    const int b = gg / S, s = gg % S;
    out_points[((size_t)b * 320 + chofs + c) * S + s] = m;
  }
}

// ---------------------------------------------------------------------------
// Host side
// ---------------------------------------------------------------------------
extern "C" void kernel_launch(void* const* d_in, const int* in_sizes, int n_in,
                              void* d_out, int out_size, void* d_ws, size_t ws_size,
                              hipStream_t stream) {
  (void)in_sizes; (void)n_in; (void)out_size; (void)ws_size;

  const float* xyz    = (const float*)d_in[0];
  const float* points = (const float*)d_in[1];
  // params flattened: per branch, per layer: w, b, gamma, beta, mean, var
  auto P = [&](int br, int ly, int f) -> const float* {
    return (const float*)d_in[2 + (br * 3 + ly) * 6 + f];
  };

  const int MLP[3][3]  = {{32, 32, 64}, {64, 64, 128}, {64, 96, 128}};
  const int KS[3]      = {16, 32, 64};
  const float R2[3]    = {0.01f, 0.04f, 0.16f};

  char*  base = (char*)d_ws;
  size_t off  = 0;
  auto alloc = [&](size_t bytes) -> void* {
    off = (off + 255) & ~(size_t)255;
    void* p = base + off;
    off += bytes;
    return p;
  };

  int*   fps_idx = (int*)  alloc((size_t)B * S * sizeof(int));
  float* newxyz  = (float*)alloc((size_t)B * S * 3 * sizeof(float));
  int* gidx[3];
  for (int i = 0; i < 3; ++i) gidx[i] = (int*)alloc((size_t)B * S * KS[i] * sizeof(int));

  _Float16* wpad[3][3];
  float*    scl[3][3];
  float*    shf[3][3];
  int kpad[3][3], cin_[3][3];
  for (int br = 0; br < 3; ++br) {
    int cin = CIN;
    for (int ly = 0; ly < 3; ++ly) {
      const int cout = MLP[br][ly];
      const int kp   = (ly == 0) ? K1PAD : cin;   // cin multiple of 32 for ly>0
      kpad[br][ly] = kp;
      cin_[br][ly] = cin;
      wpad[br][ly] = (_Float16*)alloc((size_t)kp * cout * sizeof(_Float16));
      scl[br][ly]  = (float*)alloc((size_t)cout * sizeof(float));
      shf[br][ly]  = (float*)alloc((size_t)cout * sizeof(float));
      cin = cout;
    }
  }

  float* out_xyz    = (float*)d_out;                       // (B,3,S)
  float* out_points = out_xyz + (size_t)B * 3 * S;         // (B,320,S)

  // 1) fold BN + pad/convert/swizzle weights into B-fragment layout
  for (int br = 0; br < 3; ++br)
    for (int ly = 0; ly < 3; ++ly)
      prep_layer_kernel<<<48, 256, 0, stream>>>(
          P(br, ly, 0), P(br, ly, 1), P(br, ly, 2), P(br, ly, 3), P(br, ly, 4),
          P(br, ly, 5), cin_[br][ly], MLP[br][ly], kpad[br][ly],
          wpad[br][ly], scl[br][ly], shf[br][ly]);

  // 2) farthest point sampling (+ new_xyz outputs)
  fps_kernel<<<B, 256, 0, stream>>>(xyz, fps_idx, newxyz, out_xyz);

  // 3) ball queries
  ballq_kernel<16><<<(B * S) / 256, 256, 0, stream>>>(xyz, newxyz, gidx[0], R2[0]);
  ballq_kernel<32><<<(B * S) / 256, 256, 0, stream>>>(xyz, newxyz, gidx[1], R2[1]);
  ballq_kernel<64><<<(B * S) / 256, 256, 0, stream>>>(xyz, newxyz, gidx[2], R2[2]);

  // 4) fused gather + MLP (WMMA) + max-pool, per branch
  msg_branch_kernel<16, 32, 32, 64><<<(B * S) / (ROWS / 16), 256, 0, stream>>>(
      xyz, points, newxyz, gidx[0], wpad[0][0], wpad[0][1], wpad[0][2],
      scl[0][0], shf[0][0], scl[0][1], shf[0][1], scl[0][2], shf[0][2],
      out_points, 0);
  msg_branch_kernel<32, 64, 64, 128><<<(B * S) / (ROWS / 32), 256, 0, stream>>>(
      xyz, points, newxyz, gidx[1], wpad[1][0], wpad[1][1], wpad[1][2],
      scl[1][0], shf[1][0], scl[1][1], shf[1][1], scl[1][2], shf[1][2],
      out_points, 64);
  msg_branch_kernel<64, 64, 96, 128><<<(B * S) / (ROWS / 64), 256, 0, stream>>>(
      xyz, points, newxyz, gidx[2], wpad[2][0], wpad[2][1], wpad[2][2],
      scl[2][0], shf[2][0], scl[2][1], shf[2][1], scl[2][2], shf[2][2],
      out_points, 192);
}
